// MultiHeadSelfAttention_63316407878099
// MI455X (gfx1250) — matrix operand order repaired and checked by reading, not compiled, not run on hardware
//
#include <hip/hip_runtime.h>

// MI455X / gfx1250 MHA forward.
// Compute-bound (~137 GFLOP vs ~100 MB traffic @ 23.3 TB/s) -> everything
// through v_wmma_f32_16x16x32_bf16, flash-style online softmax so scores
// never hit HBM. GEMM stages the shared W tile into LDS via the CDNA5
// async-to-LDS path (ASYNCcnt) so the 4 waves of a block don't re-read it.

#define D_MODEL 1024
#define N_HEAD  16
#define D_HEAD  64
#define BATCH   4
#define SEQ     2048
#define BS      (BATCH * SEQ)
#define WSTRIDE (D_MODEL + 8)   // LDS row pad: 16 lanes -> 16 distinct banks

#if __has_builtin(__builtin_amdgcn_global_load_async_to_lds_b128)
#define HAVE_ASYNC_LDS 1
#else
#define HAVE_ASYNC_LDS 0
#endif

typedef __attribute__((ext_vector_type(16))) __bf16 v16bf;
typedef __attribute__((ext_vector_type(8)))  float  v8f;
typedef int v4i __attribute__((vector_size(16)));
typedef __attribute__((address_space(1))) v4i* gptr_v4i;
typedef __attribute__((address_space(3))) v4i* lptr_v4i;

union FragB {
  v16bf v;
  uint4 q[2];
};

__device__ __forceinline__ __bf16 f2bf(float f) {
  union { float f; unsigned u; } c; c.f = f;
  unsigned r = c.u + 0x7fffu + ((c.u >> 16) & 1u);   // round-to-nearest-even
  unsigned short h = (unsigned short)(r >> 16);
  union { unsigned short s; __bf16 b; } o; o.s = h;
  return o.b;
}

__global__ void cvt_f32_bf16(const float* __restrict__ src,
                             __bf16* __restrict__ dst, int n) {
  int i = blockIdx.x * blockDim.x + threadIdx.x;
  int stride = gridDim.x * blockDim.x;
  for (; i < n; i += stride) dst[i] = f2bf(src[i]);
}

// Y[M, N=1024] = A[M, K=1024] @ W[N, K]^T + bias
// Block = 4 waves sharing one 16-col W strip (staged in LDS); each wave
// computes a 16x16 tile over the full K.
// mode 0/1: write bf16 to [B,H,S,Dh] (Q / K)
// mode 2  : write bf16 to [B,H,Dh,S] (V transposed)
// mode 3  : write f32 row-major [M, N] (final output projection)
__global__ void __launch_bounds__(128)
gemm_bf16(const __bf16* __restrict__ A, const __bf16* __restrict__ W,
          const float* __restrict__ bias, __bf16* __restrict__ outb,
          float* __restrict__ outf, int mode) {
  __shared__ __bf16 Wlds[16 * WSTRIDE];   // ~33 KB of the WGP's 320 KB

  const int K = D_MODEL;
  int lane = threadIdx.x & 31;
  int wave = threadIdx.x >> 5;
  int lr = lane & 15;
  int hi = lane >> 4;
  int m0 = (blockIdx.y * 4 + wave) * 16;
  int n0 = blockIdx.x * 16;

  // Cooperatively stage the 16 x 1024 W strip (32 KB) into LDS.
  // 2048 chunks of 16 B; 128 threads -> 16 async b128 transfers each.
  for (int i = threadIdx.x; i < 16 * (K / 8); i += 128) {
    int row = i >> 7;
    int col = (i & 127) * 8;
    const __bf16* g = W + (size_t)(n0 + row) * K + col;
    __bf16* l = &Wlds[row * WSTRIDE + col];
#if HAVE_ASYNC_LDS
    __builtin_amdgcn_global_load_async_to_lds_b128((gptr_v4i)g, (lptr_v4i)l,
                                                   0, 0);
#else
    *(uint4*)l = *(const uint4*)g;
#endif
  }
#if HAVE_ASYNC_LDS
#if __has_builtin(__builtin_amdgcn_s_wait_asynccnt)
  __builtin_amdgcn_s_wait_asynccnt(0);
#else
  asm volatile("s_wait_asynccnt 0x0" ::: "memory");
#endif
#endif
  __syncthreads();

  const __bf16* arow = A + (size_t)(m0 + lr) * K;       // A-frag: lane = row M
  const __bf16* bldr = &Wlds[lr * WSTRIDE];             // B-frag: lane = col N

  v8f acc = {};
#pragma unroll 4
  for (int k = 0; k < K; k += 32) {
    __builtin_prefetch(arow + k + 256, 0, 1);           // global_prefetch_b8
    FragB a, b;
    a.q[0] = *(const uint4*)(arow + k + hi * 16);       // K 0..7 / 16..23
    a.q[1] = *(const uint4*)(arow + k + hi * 16 + 8);   // K 8..15 / 24..31
    b.q[0] = *(const uint4*)(bldr + k + hi * 16);       // K 0..15 / 16..31
    b.q[1] = *(const uint4*)(bldr + k + hi * 16 + 8);
    acc = __builtin_amdgcn_wmma_f32_16x16x32_bf16(false, a.v, false, b.v,
                                                  (short)0, acc, false, false);
  }

  int e = n0 + lr;
  float bcol = bias[e];
  int h = e >> 6, dh = e & 63;
#pragma unroll
  for (int r = 0; r < 8; ++r) {
    int m = m0 + r + hi * 8;       // C layout: row = r (+8 for high half)
    int b = m >> 11;               // / SEQ
    int s = m & (SEQ - 1);
    float y = acc[r] + bcol;
    if (mode == 3) {
      outf[(size_t)m * D_MODEL + e] = y;
    } else if (mode == 2) {
      outb[((size_t)(b * N_HEAD + h) * D_HEAD + dh) * SEQ + s] = f2bf(y);
    } else {
      outb[((size_t)(b * N_HEAD + h) * SEQ + s) * D_HEAD + dh] = f2bf(y);
    }
  }
}

// Flash-style causal attention. One wave per 16-query strip of one (b,h).
// Q,K: [B,H,S,64] bf16.  Vt: [B,H,64,S] bf16.  Ctx out: [B,S,D] bf16.
__global__ void __launch_bounds__(128)
attn_fwd(const __bf16* __restrict__ Q, const __bf16* __restrict__ Kc,
         const __bf16* __restrict__ Vt, __bf16* __restrict__ Ctx) {
  __shared__ __bf16 Plds[4][16 * 32];   // per-wave 16x32 P tile (C->A relayout)

  int lane = threadIdx.x & 31;
  int wave = threadIdx.x >> 5;
  int lr = lane & 15;
  int hi = lane >> 4;
  int bh = blockIdx.x;
  int qt = blockIdx.y * 4 + wave;

  const __bf16* Qb = Q  + (size_t)bh * SEQ * D_HEAD;
  const __bf16* Kb = Kc + (size_t)bh * SEQ * D_HEAD;
  const __bf16* Vb = Vt + (size_t)bh * D_HEAD * SEQ;

  // Q A-fragments for d chunks [0,32) and [32,64) -- loaded once.
  const __bf16* qrow = Qb + (size_t)(qt * 16 + lr) * D_HEAD;
  FragB qa0, qa1;
  qa0.q[0] = *(const uint4*)(qrow + hi * 16);
  qa0.q[1] = *(const uint4*)(qrow + hi * 16 + 8);
  qa1.q[0] = *(const uint4*)(qrow + 32 + hi * 16);
  qa1.q[1] = *(const uint4*)(qrow + 32 + hi * 16 + 8);

  float mrow[8], lrow[8];
  v8f acc[4];
  v8f zero = {};
#pragma unroll
  for (int r = 0; r < 8; ++r) { mrow[r] = -1e30f; lrow[r] = 0.f; }
#pragma unroll
  for (int d = 0; d < 4; ++d) acc[d] = zero;

  const float scale = 0.125f;  // 1/sqrt(64)
  int nkb = (qt * 16 + 15) / 32 + 1;   // causal: only key blocks <= query

  for (int kb = 0; kb < nkb; ++kb) {
    int k0 = kb * 32;

    // scores: two 16-key tiles, K-dim = d (64 = 2 WMMA each)
    v8f s0 = zero, s1 = zero;
    {
      const __bf16* kr = Kb + (size_t)(k0 + lr) * D_HEAD;
      FragB b0, b1;
      b0.q[0] = *(const uint4*)(kr + hi * 16);
      b0.q[1] = *(const uint4*)(kr + hi * 16 + 8);
      b1.q[0] = *(const uint4*)(kr + 32 + hi * 16);
      b1.q[1] = *(const uint4*)(kr + 32 + hi * 16 + 8);
      s0 = __builtin_amdgcn_wmma_f32_16x16x32_bf16(false, qa0.v, false, b0.v,
                                                   (short)0, s0, false, false);
      s0 = __builtin_amdgcn_wmma_f32_16x16x32_bf16(false, qa1.v, false, b1.v,
                                                   (short)0, s0, false, false);
    }
    {
      const __bf16* kr = Kb + (size_t)(k0 + 16 + lr) * D_HEAD;
      FragB b0, b1;
      b0.q[0] = *(const uint4*)(kr + hi * 16);
      b0.q[1] = *(const uint4*)(kr + hi * 16 + 8);
      b1.q[0] = *(const uint4*)(kr + 32 + hi * 16);
      b1.q[1] = *(const uint4*)(kr + 32 + hi * 16 + 8);
      s1 = __builtin_amdgcn_wmma_f32_16x16x32_bf16(false, qa0.v, false, b0.v,
                                                   (short)0, s1, false, false);
      s1 = __builtin_amdgcn_wmma_f32_16x16x32_bf16(false, qa1.v, false, b1.v,
                                                   (short)0, s1, false, false);
    }

    // scale + causal mask + block row-max (row M=r+hi*8 lives on one 16-lane half)
    float bm[8];
#pragma unroll
    for (int r = 0; r < 8; ++r) {
      int qi = qt * 16 + r + hi * 8;
      float v0 = s0[r] * scale; if (k0 + lr > qi)      v0 = -1e9f;
      float v1 = s1[r] * scale; if (k0 + 16 + lr > qi) v1 = -1e9f;
      s0[r] = v0; s1[r] = v1;
      bm[r] = fmaxf(v0, v1);
    }
#pragma unroll
    for (int off = 1; off < 16; off <<= 1)
#pragma unroll
      for (int r = 0; r < 8; ++r)
        bm[r] = fmaxf(bm[r], __shfl_xor(bm[r], off, 32));

    // online softmax update; stash P (bf16) into LDS in C layout
#pragma unroll
    for (int r = 0; r < 8; ++r) {
      float nm = fmaxf(mrow[r], bm[r]);
      float sc = __expf(mrow[r] - nm);
      mrow[r] = nm;
      float p0 = __expf(s0[r] - nm);
      float p1 = __expf(s1[r] - nm);
      float rs = p0 + p1;
#pragma unroll
      for (int off = 1; off < 16; off <<= 1) rs += __shfl_xor(rs, off, 32);
      lrow[r] = lrow[r] * sc + rs;
#pragma unroll
      for (int d = 0; d < 4; ++d) acc[d][r] = acc[d][r] * sc;
      int row = r + hi * 8;
      Plds[wave][row * 32 + lr]      = f2bf(p0);
      Plds[wave][row * 32 + 16 + lr] = f2bf(p1);
    }

    // reload P as an A-fragment (16x32, K = key-in-block)
    FragB pa;
    pa.q[0] = *(const uint4*)&Plds[wave][lr * 32 + hi * 16];
    pa.q[1] = *(const uint4*)&Plds[wave][lr * 32 + hi * 16 + 8];

    // context += P @ V : B-frag lane = d, contiguous keys from Vt
#pragma unroll
    for (int dt = 0; dt < 4; ++dt) {
      const __bf16* vr = Vb + (size_t)(dt * 16 + lr) * SEQ + k0;
      FragB bv;
      bv.q[0] = *(const uint4*)(vr + hi * 16);
      bv.q[1] = *(const uint4*)(vr + hi * 16 + 8);
      acc[dt] = __builtin_amdgcn_wmma_f32_16x16x32_bf16(false, pa.v, false, bv.v,
                                                        (short)0, acc[dt], false, false);
    }
  }

  // normalize and scatter context to [B,S,D] bf16
  int b = bh / N_HEAD, h = bh % N_HEAD;
#pragma unroll
  for (int r = 0; r < 8; ++r) {
    float inv = 1.0f / lrow[r];
    int s = qt * 16 + r + hi * 8;
    size_t base = ((size_t)(b * SEQ + s)) * D_MODEL + h * D_HEAD;
#pragma unroll
    for (int dt = 0; dt < 4; ++dt)
      Ctx[base + dt * 16 + lr] = f2bf(acc[dt][r] * inv);
  }
}

extern "C" void kernel_launch(void* const* d_in, const int* in_sizes, int n_in,
                              void* d_out, int out_size, void* d_ws, size_t ws_size,
                              hipStream_t stream) {
  (void)in_sizes; (void)n_in; (void)out_size; (void)ws_size;
  const float* x  = (const float*)d_in[0];
  const float* Wq = (const float*)d_in[1];
  const float* bq = (const float*)d_in[2];
  const float* Wk = (const float*)d_in[3];
  const float* bk = (const float*)d_in[4];
  const float* Wv = (const float*)d_in[5];
  const float* bv = (const float*)d_in[6];
  const float* Wo = (const float*)d_in[7];
  const float* bo = (const float*)d_in[8];
  // d_in[9] = mask: causal, handled analytically.
  float* out = (float*)d_out;

  char* ws = (char*)d_ws;
  __bf16* Xb   = (__bf16*)ws; ws += (size_t)BS * D_MODEL * 2;        // 16 MB
  __bf16* Wqb  = (__bf16*)ws; ws += (size_t)D_MODEL * D_MODEL * 2;   //  2 MB
  __bf16* Wkb  = (__bf16*)ws; ws += (size_t)D_MODEL * D_MODEL * 2;
  __bf16* Wvb  = (__bf16*)ws; ws += (size_t)D_MODEL * D_MODEL * 2;
  __bf16* Wob  = (__bf16*)ws; ws += (size_t)D_MODEL * D_MODEL * 2;
  __bf16* Qb   = (__bf16*)ws; ws += (size_t)BS * D_MODEL * 2;        // [B,H,S,Dh]
  __bf16* Kb   = (__bf16*)ws; ws += (size_t)BS * D_MODEL * 2;        // [B,H,S,Dh]
  __bf16* Vtb  = (__bf16*)ws; ws += (size_t)BS * D_MODEL * 2;        // [B,H,Dh,S]
  __bf16* Ctxb = (__bf16*)ws; ws += (size_t)BS * D_MODEL * 2;        // [B,S,D]

  cvt_f32_bf16<<<512, 256, 0, stream>>>(x,  Xb,  BS * D_MODEL);
  cvt_f32_bf16<<<512, 256, 0, stream>>>(Wq, Wqb, D_MODEL * D_MODEL);
  cvt_f32_bf16<<<512, 256, 0, stream>>>(Wk, Wkb, D_MODEL * D_MODEL);
  cvt_f32_bf16<<<512, 256, 0, stream>>>(Wv, Wvb, D_MODEL * D_MODEL);
  cvt_f32_bf16<<<512, 256, 0, stream>>>(Wo, Wob, D_MODEL * D_MODEL);

  dim3 gg(D_MODEL / 16, BS / 64);   // 64 x 128 blocks, 4 waves each
  gemm_bf16<<<gg, 128, 0, stream>>>(Xb, Wqb, bq, Qb,  nullptr, 0);
  gemm_bf16<<<gg, 128, 0, stream>>>(Xb, Wkb, bk, Kb,  nullptr, 1);
  gemm_bf16<<<gg, 128, 0, stream>>>(Xb, Wvb, bv, Vtb, nullptr, 2);

  attn_fwd<<<dim3(BATCH * N_HEAD, SEQ / 64), 128, 0, stream>>>(Qb, Kb, Vtb, Ctxb);

  gemm_bf16<<<gg, 128, 0, stream>>>(Ctxb, Wob, bo, nullptr, out, 3);
}
